// MaskedCausalAttention_68796786148003
// MI455X (gfx1250) — compile-verified, matrix operand order checked by reading, not run on hardware
//
#include <hip/hip_runtime.h>
#include <hip/hip_bf16.h>

// ---------------------------------------------------------------------------
// Problem constants (from reference): B=4, T=1024, C=1024, N_HEADS=16, D=64
// ---------------------------------------------------------------------------
#define BATCH 4
#define TSEQ  1024
#define CDIM  1024
#define NH    16
#define DH    64
#define MROWS (BATCH * TSEQ)   // 4096

typedef __attribute__((ext_vector_type(16))) __bf16 v16bf;
typedef __attribute__((ext_vector_type(8)))  __bf16 v8bf;
typedef __attribute__((ext_vector_type(8)))  float  v8f;
typedef __attribute__((ext_vector_type(4)))  unsigned int v4u;
typedef __attribute__((ext_vector_type(8)))  int v8i;
typedef __attribute__((ext_vector_type(4)))  int v4i;

union V16U { v16bf v; v8bf h[2]; };

static __device__ __forceinline__ v8bf ld8(const __bf16* p) {
  return *reinterpret_cast<const v8bf*>(p);
}
static __device__ __forceinline__ v16bf mk16(v8bf lo, v8bf hi) {
  V16U u; u.h[0] = lo; u.h[1] = hi; return u.v;
}

// async global -> LDS copy of one 16-byte chunk (per lane), ASYNCcnt-tracked
static __device__ __forceinline__ void async_cp16(const __bf16* gsrc, const __bf16* ldst) {
  unsigned lds_addr = (unsigned)(size_t)(const void*)ldst;
  unsigned long long gaddr = (unsigned long long)(size_t)(const void*)gsrc;
  asm volatile("global_load_async_to_lds_b128 %0, %1, off"
               :: "v"(lds_addr), "v"(gaddr) : "memory");
}
static __device__ __forceinline__ void wait_async0() {
  asm volatile("s_wait_asynccnt 0x0" ::: "memory");
}
static __device__ __forceinline__ void wait_tensor0() {
#if __has_builtin(__builtin_amdgcn_s_wait_tensorcnt)
  __builtin_amdgcn_s_wait_tensorcnt((short)0);
#else
  asm volatile("s_wait_tensorcnt 0x0" ::: "memory");
#endif
}

// ---------------------------------------------------------------------------
// fp32 -> bf16 convert (copy)
// ---------------------------------------------------------------------------
__global__ void mca_cvt_bf16(const float* __restrict__ in,
                             __bf16* __restrict__ out, int n) {
  int i = blockIdx.x * blockDim.x + threadIdx.x;
  if (i < n) out[i] = (__bf16)in[i];
}

// ---------------------------------------------------------------------------
// fp32 [K,N] -> bf16 transposed [N,K]   (WT[n*K+k] = W[k*N+n])
// ---------------------------------------------------------------------------
__global__ void mca_transpose_bf16(const float* __restrict__ W,
                                   __bf16* __restrict__ WT,
                                   int K, int N) {
  int i = blockIdx.x * blockDim.x + threadIdx.x;
  if (i < K * N) {
    int k = i / N, n = i % N;
    WT[(size_t)n * K + k] = (__bf16)W[(size_t)k * N + n];
  }
}

// ---------------------------------------------------------------------------
// bf16 GEMM  Y = X @ W + bias, LDS-staged via async copies + TDM.
//   X  : [M, 1024] bf16 row-major
//   WT : [1024, 1024] bf16 = W^T row-major  (WMMA B operand loads)
//   mode 0: out_f [M,1024] fp32 (final projection -> d_out)
//   mode 1: out_bf as [B, NH, T, DH]  (Q, K)
//   mode 2: out_bf as [B, NH, DH, T]  (V transposed per head)
// Block = 128 threads (4 waves) computing a 64x64 tile; per 32-wide k-step the
// A panel (64x32) is async-copied to LDS by all lanes and the block-uniform B
// panel (64x32) is DMA'd by the Tensor Data Mover from wave 0.
// Grid = (M/64, 1024/64).
// ---------------------------------------------------------------------------
__global__ void mca_gemm_bf16(const __bf16* __restrict__ X,
                              const __bf16* __restrict__ WT,
                              const float*  __restrict__ bias,
                              int mode,
                              __bf16* __restrict__ out_bf,
                              float*  __restrict__ out_f) {
  __shared__ __bf16 shA[64 * 32];   // A panel: 64 rows x 32 k  (4 KiB)
  __shared__ __bf16 shB[64 * 32];   // B panel: 64 cols x 32 k  (4 KiB)

  const int tid  = threadIdx.x;      // 0..127
  const int lane = tid & 31;
  const int wave = tid >> 5;
  const int hl   = lane >> 4;        // half-wave selector (K-interleave)
  const int lr   = lane & 15;        // row/col within 16

  const int mb4 = blockIdx.x;        // 64-row block
  const int nb  = blockIdx.y;        // 64-col block
  const int Kd  = CDIM;

  v8f acc0 = {}, acc1 = {}, acc2 = {}, acc3 = {};

  for (int k = 0; k < Kd; k += 32) {
    __syncthreads();   // previous k-step's LDS reads complete

    // ---- stage A panel: 64 rows x 64 bytes = 256 x 16B chunks, 2/thread ----
#pragma unroll
    for (int i = 0; i < 2; ++i) {
      const int c   = tid * 2 + i;         // chunk id 0..255
      const int row = c >> 2;              // 0..63
      const int cc  = (c & 3) * 8;         // element offset in row (0,8,16,24)
      async_cp16(X + (size_t)(mb4 * 64 + row) * Kd + k + cc,
                 shA + row * 32 + cc);
    }

    // ---- stage B panel ----
#if __has_builtin(__builtin_amdgcn_tensor_load_to_lds)
    if (wave == 0) {
      // TDM descriptor: 2D tensor (row len 1024 bf16), tile 32(k) x 64(rows)
      const size_t gaddr = (size_t)(WT + (size_t)(nb * 64) * Kd + k);
      const unsigned lds_b = (unsigned)(size_t)(const void*)shB;
      v4u g0 = { 1u,                                    // count=1, user D#
                 lds_b,                                 // lds_addr
                 (unsigned)(gaddr & 0xFFFFFFFFu),       // global_addr[31:0]
                 (unsigned)((gaddr >> 32) & 0x01FFFFFFu) | (2u << 30) }; // [56:32] | type=2
      v8i g1 = { (int)0x00010000,                       // data_size=2B, mask=0
                 (int)(1024u << 16),                    // tensor_dim0=1024 (lo16 -> [31:16])
                 0,                                     // td0 hi16=0, td1 lo16=0
                 (int)(16u | (32u << 16)),              // td1 hi16 (td1=1<<20), tile_dim0=32
                 (int)64,                               // tile_dim1=64, tile_dim2=0
                 (int)1024,                             // tensor_dim0_stride[31:0]=1024
                 0, 0 };                                // stride hi, dim1_stride
      v4i z4 = {0, 0, 0, 0};
#if __clang_major__ >= 23
      v8i z8 = {0, 0, 0, 0, 0, 0, 0, 0};
      __builtin_amdgcn_tensor_load_to_lds(g0, g1, z4, z4, z8, 0);
#else
      __builtin_amdgcn_tensor_load_to_lds(g0, g1, z4, z4, 0);
#endif
      wait_tensor0();
    }
#else
    // fallback: async-copy B panel like A
#pragma unroll
    for (int i = 0; i < 2; ++i) {
      const int c   = tid * 2 + i;
      const int row = c >> 2;
      const int cc  = (c & 3) * 8;
      async_cp16(WT + (size_t)(nb * 64 + row) * Kd + k + cc,
                 shB + row * 32 + cc);
    }
#endif

    wait_async0();
    __syncthreads();

    // ---- compute: wave w does rows w*16..w*16+15 against 4 B tiles ----
    const __bf16* arow = shA + (wave * 16 + lr) * 32;
    v16bf a = mk16(ld8(arow + hl * 8), ld8(arow + 16 + hl * 8));
    const __bf16* b0r = shB + ( 0 + lr) * 32;
    const __bf16* b1r = shB + (16 + lr) * 32;
    const __bf16* b2r = shB + (32 + lr) * 32;
    const __bf16* b3r = shB + (48 + lr) * 32;
    v16bf b0 = mk16(ld8(b0r + hl * 8), ld8(b0r + 16 + hl * 8));
    v16bf b1 = mk16(ld8(b1r + hl * 8), ld8(b1r + 16 + hl * 8));
    v16bf b2 = mk16(ld8(b2r + hl * 8), ld8(b2r + 16 + hl * 8));
    v16bf b3 = mk16(ld8(b3r + hl * 8), ld8(b3r + 16 + hl * 8));
    acc0 = __builtin_amdgcn_wmma_f32_16x16x32_bf16(false, a, false, b0, (short)0, acc0, false, false);
    acc1 = __builtin_amdgcn_wmma_f32_16x16x32_bf16(false, a, false, b1, (short)0, acc1, false, false);
    acc2 = __builtin_amdgcn_wmma_f32_16x16x32_bf16(false, a, false, b2, (short)0, acc2, false, false);
    acc3 = __builtin_amdgcn_wmma_f32_16x16x32_bf16(false, a, false, b3, (short)0, acc3, false, false);
  }

  const int mb = mb4 * 4 + wave;
  v8f accs[4] = {acc0, acc1, acc2, acc3};
#pragma unroll
  for (int j = 0; j < 4; ++j) {
#pragma unroll
    for (int r = 0; r < 8; ++r) {
      const int mg  = mb * 16 + r + hl * 8;       // global row in [0, M)
      const int col = nb * 64 + j * 16 + lr;      // global col in [0, 1024)
      const float v = accs[j][r] + bias[col];
      if (mode == 0) {
        out_f[(size_t)mg * CDIM + col] = v;
      } else {
        const int b = mg >> 10, t = mg & (TSEQ - 1);
        const int h = col >> 6, d = col & (DH - 1);
        if (mode == 1)
          out_bf[(((size_t)(b * NH + h) * TSEQ) + t) * DH + d] = (__bf16)v;
        else
          out_bf[(((size_t)(b * NH + h) * DH) + d) * TSEQ + t] = (__bf16)v;
      }
    }
  }
}

// ---------------------------------------------------------------------------
// Attention core: one wave per (b, head, 16-row tile of T).
//   Q,K : [B, NH, T, DH] bf16 ;  Vt : [B, NH, DH, T] bf16
//   AO  : [B, T, C] bf16  (attention output, feeds final projection)
// scores = (Q K^T)/8 + (-1e9 on s<=t) [faithful lower-tri mask], softmax(s),
// then P @ V. Score strip (16x1024) lives in LDS; P bf16 strip reloaded in
// WMMA A-layout for the P@V pass.
// Dynamic LDS: 16*1024*4 (f32 scores) + 16*1024*2 (bf16 P) = 96 KiB.
// ---------------------------------------------------------------------------
__global__ void mca_attn(const __bf16* __restrict__ Q,
                         const __bf16* __restrict__ K,
                         const __bf16* __restrict__ Vt,
                         __bf16* __restrict__ AO) {
  extern __shared__ char smem[];
  float*  sh_s = (float*)smem;                         // [16][1024] f32
  __bf16* sh_p = (__bf16*)(smem + 16 * TSEQ * 4);      // [16][1024] bf16

  const int wg   = blockIdx.x;            // B*NH*(T/16) = 4096
  const int tb   = (wg & 63) * 16;        // row-tile base in T
  const int head = (wg >> 6) & (NH - 1);
  const int b    = wg >> 10;
  const int lane = threadIdx.x;
  const int hl   = lane >> 4;
  const int lr   = lane & 15;

  const __bf16* Qh = Q  + ((size_t)(b * NH + head) * TSEQ) * DH;
  const __bf16* Kh = K  + ((size_t)(b * NH + head) * TSEQ) * DH;
  const __bf16* Vh = Vt + ((size_t)(b * NH + head) * DH) * TSEQ;

  // --- load Q tile in A-layout (two 16x32 chunks over D=64) ---
  const __bf16* qrow = Qh + (size_t)(tb + lr) * DH;
  v16bf qa0 = mk16(ld8(qrow +  0 + hl * 8), ld8(qrow + 16 + hl * 8));
  v16bf qa1 = mk16(ld8(qrow + 32 + hl * 8), ld8(qrow + 48 + hl * 8));

  // --- phase 1: full score strip, scaled + masked, into LDS ---
  for (int sb = 0; sb < TSEQ; sb += 16) {
    const __bf16* krow = Kh + (size_t)(sb + lr) * DH;   // B-tile: col = sb+lr
    __builtin_prefetch(krow + 16 * DH, 0, 0);           // global_prefetch_b8
    v16bf kb0 = mk16(ld8(krow +  0 + hl * 8), ld8(krow + 16 + hl * 8));
    v16bf kb1 = mk16(ld8(krow + 32 + hl * 8), ld8(krow + 48 + hl * 8));
    v8f acc = {};
    acc = __builtin_amdgcn_wmma_f32_16x16x32_bf16(false, qa0, false, kb0, (short)0, acc, false, false);
    acc = __builtin_amdgcn_wmma_f32_16x16x32_bf16(false, qa1, false, kb1, (short)0, acc, false, false);
    const int scol = sb + lr;
#pragma unroll
    for (int r = 0; r < 8; ++r) {
      const int trow = tb + r + hl * 8;
      float v = acc[r] * 0.125f;                 // 1/sqrt(64)
      if (scol <= trow) v += -1e9f;              // faithful lower-tri mask
      sh_s[(r + hl * 8) * TSEQ + scol] = v;
    }
  }
  __syncthreads();

  // --- phase 2: softmax per row (wave-cooperative, shfl_xor reductions) ---
  for (int r = 0; r < 16; ++r) {
    const float* row = sh_s + r * TSEQ;
    float m = -3.0e38f;
    for (int c = lane; c < TSEQ; c += 32) m = fmaxf(m, row[c]);
#pragma unroll
    for (int off = 16; off > 0; off >>= 1) m = fmaxf(m, __shfl_xor(m, off, 32));
    float s = 0.0f;
    for (int c = lane; c < TSEQ; c += 32) s += __expf(row[c] - m);
#pragma unroll
    for (int off = 16; off > 0; off >>= 1) s += __shfl_xor(s, off, 32);
    const float inv = 1.0f / s;
    for (int c = lane; c < TSEQ; c += 32)
      sh_p[r * TSEQ + c] = (__bf16)(__expf(row[c] - m) * inv);
  }
  __syncthreads();

  // --- phase 3: O = P @ V  (K-dim = T, chunks of 32) ---
  v8f o0 = {}, o1 = {}, o2 = {}, o3 = {};
  const __bf16* prow = sh_p + (size_t)lr * TSEQ;
  const __bf16* v0 = Vh + (size_t)( 0 + lr) * TSEQ;
  const __bf16* v1 = Vh + (size_t)(16 + lr) * TSEQ;
  const __bf16* v2 = Vh + (size_t)(32 + lr) * TSEQ;
  const __bf16* v3 = Vh + (size_t)(48 + lr) * TSEQ;
  for (int sc = 0; sc < TSEQ; sc += 32) {
    v16bf pa = mk16(ld8(prow + sc + hl * 8), ld8(prow + sc + 16 + hl * 8));
    v16bf vb0 = mk16(ld8(v0 + sc + hl * 8), ld8(v0 + sc + 16 + hl * 8));
    v16bf vb1 = mk16(ld8(v1 + sc + hl * 8), ld8(v1 + sc + 16 + hl * 8));
    v16bf vb2 = mk16(ld8(v2 + sc + hl * 8), ld8(v2 + sc + 16 + hl * 8));
    v16bf vb3 = mk16(ld8(v3 + sc + hl * 8), ld8(v3 + sc + 16 + hl * 8));
    o0 = __builtin_amdgcn_wmma_f32_16x16x32_bf16(false, pa, false, vb0, (short)0, o0, false, false);
    o1 = __builtin_amdgcn_wmma_f32_16x16x32_bf16(false, pa, false, vb1, (short)0, o1, false, false);
    o2 = __builtin_amdgcn_wmma_f32_16x16x32_bf16(false, pa, false, vb2, (short)0, o2, false, false);
    o3 = __builtin_amdgcn_wmma_f32_16x16x32_bf16(false, pa, false, vb3, (short)0, o3, false, false);
  }

  v8f os[4] = {o0, o1, o2, o3};
#pragma unroll
  for (int j = 0; j < 4; ++j) {
#pragma unroll
    for (int r = 0; r < 8; ++r) {
      const size_t row = (size_t)b * TSEQ + tb + r + hl * 8;
      const int    col = head * DH + j * 16 + lr;
      AO[row * CDIM + col] = (__bf16)os[j][r];
    }
  }
}

// ---------------------------------------------------------------------------
// Host-side launcher
// ---------------------------------------------------------------------------
extern "C" void kernel_launch(void* const* d_in, const int* in_sizes, int n_in,
                              void* d_out, int out_size, void* d_ws, size_t ws_size,
                              hipStream_t stream) {
  const float* x  = (const float*)d_in[0];
  const float* Wq = (const float*)d_in[1];
  const float* bq = (const float*)d_in[2];
  const float* Wk = (const float*)d_in[3];
  const float* bk = (const float*)d_in[4];
  const float* Wv = (const float*)d_in[5];
  const float* bv = (const float*)d_in[6];
  const float* Wo = (const float*)d_in[7];
  const float* bo = (const float*)d_in[8];
  float* out = (float*)d_out;

  char* ws = (char*)d_ws;
  const size_t SZ_X  = (size_t)MROWS * CDIM * 2;   // 8 MiB bf16
  const size_t SZ_W  = (size_t)CDIM * CDIM * 2;    // 2 MiB bf16
  const size_t SZ_H  = (size_t)BATCH * NH * TSEQ * DH * 2; // 8 MiB bf16
  __bf16* Xbf = (__bf16*)(ws);
  __bf16* WqT = (__bf16*)(ws + SZ_X);
  __bf16* WkT = (__bf16*)(ws + SZ_X + SZ_W);
  __bf16* WvT = (__bf16*)(ws + SZ_X + 2 * SZ_W);
  __bf16* WoT = (__bf16*)(ws + SZ_X + 3 * SZ_W);
  __bf16* Qb  = (__bf16*)(ws + SZ_X + 4 * SZ_W);
  __bf16* Kb  = (__bf16*)(ws + SZ_X + 4 * SZ_W + SZ_H);
  __bf16* Vtb = (__bf16*)(ws + SZ_X + 4 * SZ_W + 2 * SZ_H);
  __bf16* AO  = (__bf16*)(ws + SZ_X + 4 * SZ_W + 3 * SZ_H);

  // 1) convert x, transpose weights to bf16
  {
    const int n = MROWS * CDIM;
    mca_cvt_bf16<<<(n + 255) / 256, 256, 0, stream>>>(x, Xbf, n);
    const int nw = CDIM * CDIM;
    mca_transpose_bf16<<<(nw + 255) / 256, 256, 0, stream>>>(Wq, WqT, CDIM, CDIM);
    mca_transpose_bf16<<<(nw + 255) / 256, 256, 0, stream>>>(Wk, WkT, CDIM, CDIM);
    mca_transpose_bf16<<<(nw + 255) / 256, 256, 0, stream>>>(Wv, WvT, CDIM, CDIM);
    mca_transpose_bf16<<<(nw + 255) / 256, 256, 0, stream>>>(Wo, WoT, CDIM, CDIM);
  }

  // 2) QKV projections (WMMA GEMM, LDS-staged)
  {
    dim3 grid(MROWS / 64, CDIM / 64);
    mca_gemm_bf16<<<grid, 128, 0, stream>>>(Xbf, WqT, bq, 1, Qb,  nullptr);
    mca_gemm_bf16<<<grid, 128, 0, stream>>>(Xbf, WkT, bk, 1, Kb,  nullptr);
    mca_gemm_bf16<<<grid, 128, 0, stream>>>(Xbf, WvT, bv, 2, Vtb, nullptr);
  }

  // 3) attention core (one wave per 16-row tile per head)
  {
    const int nblocks = BATCH * NH * (TSEQ / 16);   // 4096
    const size_t lds  = (size_t)16 * TSEQ * 4 + (size_t)16 * TSEQ * 2; // 96 KiB
    mca_attn<<<nblocks, 32, lds, stream>>>(Qb, Kb, Vtb, AO);
  }

  // 4) output projection -> fp32 d_out
  {
    dim3 grid(MROWS / 64, CDIM / 64);
    mca_gemm_bf16<<<grid, 128, 0, stream>>>(AO, WoT, bo, 0, nullptr, out);
  }
}
